// LCAM_18133351923860
// MI455X (gfx1250) — compile-verified, matrix operand order checked by reading, not run on hardware
//
#include <hip/hip_runtime.h>
#include <hip/hip_bf16.h>
#include <math.h>

// ---------------------------------------------------------------------------
// LCAM (window attention + global token + dwconv pos branch) for gfx1250.
// B=2, H=W=256, C=192, heads=6, dh=32, window 8x8 (T=64). 256%8==0 -> the
// reference's reflect pad is a no-op and is skipped.
// All GEMMs run as bf16 V_WMMA_F32_16X16X32_BF16 with f32 accumulation; the
// whole per-window pipeline (QKV -> attention -> proj) lives in WGP LDS.
// Window inputs are staged via GLOBAL_LOAD_ASYNC_TO_LDS_B128 (ASYNCcnt path);
// weight slabs are prefetched one slab ahead with global_prefetch.
// ---------------------------------------------------------------------------

#define B_     2
#define H_     256
#define W_     256
#define C_     192
#define HEADS_ 6
#define DH_    32
#define T_     64
#define NWIN_  2048            // B * (256/8) * (256/8)
#define SCALE_ 0.17677669529663687f   // 32^-0.5

typedef __attribute__((ext_vector_type(16))) __bf16       v16bf;
typedef __attribute__((ext_vector_type(8)))  float        v8f;
typedef __attribute__((ext_vector_type(8)))  unsigned int v8u;

// ---- LDS layout for lcam_attn (dynamic shared, 309504 bytes total) --------
#define OFF_XS    0          // bf16 [64][200]   x window tile
#define OFF_GS    25600      // bf16 [64][200]   1+sigmoid(illu) gates
#define OFF_PB    0          // bf16 [6][64][66] softmax probs (aliases XS+GS)
#define OFF_QKV   51200      // bf16 [64][584]   q|k|v per token
#define OFF_WT    125952     // bf16 [576][40]   weight slab, [N][K] layout
#define OFF_SB    172032     // f32  [6][64][68] attention logits
#define OFF_RAWX  172032     // f32  [64][192]   async-staged x (aliases SB)
#define OFF_RAWI  221184     // f32  [64][192]   async-staged illu (aliases SB)
#define OFF_OB    172032     // bf16 [64][200]   attn output (aliases SB)
#define OFF_VT    276480     // bf16 [6][32][72] V transposed [c][token]
#define OFF_GKF   304128     // f32  [192]       global token
#define OFF_NQI   304896     // f32  [6][64]     1/||q||
#define OFF_NKI   306432     // f32  [6][64]     1/||k||
#define OFF_PG    307968     // f32  [6][64]     softmax prob of global key
#define SMEM_BYTES 309504

// Per-lane async DMA of 16 bytes global -> LDS (GV mode, ASYNCcnt-tracked).
// lds_off: LDS byte address (dynamic-LDS base is 0 here: no static LDS).
__device__ __forceinline__ void async_gload_b128(unsigned lds_off,
                                                 const void* gaddr) {
  asm volatile("global_load_async_to_lds_b128 %0, %1, off"
               :: "v"(lds_off), "v"(gaddr) : "memory");
}
__device__ __forceinline__ void wait_asynccnt0() {
  asm volatile("s_wait_asynccnt 0x0" ::: "memory");
}

// Load a 16x32 bf16 WMMA operand fragment from LDS.
// A operand: `base` -> row-major [M][K] tile origin, stride in bf16 elems.
// B operand: `base` -> [N][K] tile origin (K-pairs contiguous), same mapping.
// Per ISA 7.12.2: lanes 0-15 carry rows 0-15 with K pairs {0..7,16..23},
// lanes 16-31 carry rows 0-15 with K pairs {8..15,24..31}; each VGPR holds
// two consecutive K values -> 32-bit ds loads.
__device__ __forceinline__ v16bf ldfrag(const __bf16* base, int stride, int lane) {
  const int half = (lane >> 4) & 1;
  const int r    = lane & 15;
  const __bf16* p = base + r * stride;
  v8u u;
#pragma unroll
  for (int v = 0; v < 8; ++v) {
    const int kk = (v < 4) ? (half * 8 + 2 * v) : (8 + half * 8 + 2 * v);
    u[v] = *(const unsigned int*)(p + kk);
  }
  return __builtin_bit_cast(v16bf, u);
}

__device__ __forceinline__ v8f wmma_bf16(v16bf a, v16bf b, v8f c) {
  return __builtin_amdgcn_wmma_f32_16x16x32_bf16(
      /*neg_a=*/false, a, /*neg_b=*/false, b,
      /*c_mod=*/(short)0, c, /*reuse_a=*/false, /*reuse_b=*/false);
}

// ---------------------------------------------------------------------------
// Kernel 1: per-(b,c) spatial mean of x  -> ws.mean[B*C]
// ---------------------------------------------------------------------------
__global__ __launch_bounds__(256) void lcam_mean(const float* __restrict__ x,
                                                 float* __restrict__ mean) {
  __shared__ float red[256];
  const int bc = blockIdx.x;            // b*192 + c
  const int b = bc / C_, c = bc % C_;
  float s = 0.f;
  for (int p = threadIdx.x; p < H_ * W_; p += 256)
    s += x[((size_t)b * H_ * W_ + p) * C_ + c];
  red[threadIdx.x] = s;
  __syncthreads();
  for (int o = 128; o > 0; o >>= 1) {
    if (threadIdx.x < o) red[threadIdx.x] += red[threadIdx.x + o];
    __syncthreads();
  }
  if (threadIdx.x == 0) mean[bc] = red[0] * (1.0f / (float)(H_ * W_));
}

// ---------------------------------------------------------------------------
// Kernel 2: gt[b] = mean[b] @ gt_w + gt_b  (tiny, VALU)
// ---------------------------------------------------------------------------
__global__ __launch_bounds__(192) void lcam_gt(const float* __restrict__ mean,
                                               const float* __restrict__ gt_w,
                                               const float* __restrict__ gt_b,
                                               float* __restrict__ gt) {
  const int b = blockIdx.x, n = threadIdx.x;
  float s = gt_b[n];
  for (int c = 0; c < C_; ++c) s += mean[b * C_ + c] * gt_w[c * C_ + n];
  gt[b * C_ + n] = s;
}

// ---------------------------------------------------------------------------
// Kernel 3: positional branch  dwconv3x3 -> gelu(exact) -> dwconv3x3,
// fused per 16x16 tile with 2-pixel halo. Writes d_out (attention adds later).
// Conv2's SAME padding applies to the *mid* map: mid outside the image is 0.
// ---------------------------------------------------------------------------
__global__ __launch_bounds__(256) void lcam_pos(const float* __restrict__ x,
                                                const float* __restrict__ w1,
                                                const float* __restrict__ w2,
                                                float* __restrict__ out) {
  __shared__ float sin_[20 * 20];
  __shared__ float smid[18 * 18];
  const int bid  = blockIdx.x;
  const int c    = bid % C_;
  const int tile = bid / C_;
  const int tw   = tile % (W_ / 16);
  const int th   = (tile / (W_ / 16)) % (H_ / 16);
  const int b    = tile / ((W_ / 16) * (H_ / 16));
  float k1[9], k2[9];
#pragma unroll
  for (int i = 0; i < 9; ++i) { k1[i] = w1[c * 9 + i]; k2[i] = w2[c * 9 + i]; }
  const int tid = threadIdx.x;
  for (int idx = tid; idx < 400; idx += 256) {
    const int i = idx / 20, j = idx % 20;
    const int ih = th * 16 - 2 + i, iw = tw * 16 - 2 + j;
    float v = 0.f;
    if (ih >= 0 && ih < H_ && iw >= 0 && iw < W_)
      v = x[(((size_t)b * H_ + ih) * W_ + iw) * C_ + c];
    sin_[idx] = v;
  }
  __syncthreads();
  for (int idx = tid; idx < 324; idx += 256) {
    const int i = idx / 18, j = idx % 18;
    const int mh = th * 16 - 1 + i, mw = tw * 16 - 1 + j;
    float v = 0.f;
    if (mh >= 0 && mh < H_ && mw >= 0 && mw < W_) {
      float a = 0.f;
#pragma unroll
      for (int dy = 0; dy < 3; ++dy)
#pragma unroll
        for (int dx = 0; dx < 3; ++dx)
          a += sin_[(i + dy) * 20 + (j + dx)] * k1[dy * 3 + dx];
      v = 0.5f * a * (1.0f + erff(a * 0.70710678118654752f));  // exact gelu
    }
    smid[idx] = v;
  }
  __syncthreads();
  {
    const int oh = tid / 16, ow = tid % 16;
    float a = 0.f;
#pragma unroll
    for (int dy = 0; dy < 3; ++dy)
#pragma unroll
      for (int dx = 0; dx < 3; ++dx)
        a += smid[(oh + dy) * 18 + (ow + dx)] * k2[dy * 3 + dx];
    out[(((size_t)b * H_ + th * 16 + oh) * W_ + tw * 16 + ow) * C_ + c] = a;
  }
}

// ---------------------------------------------------------------------------
// Kernel 4: fused per-window QKV -> attention (+global token) -> proj.
// One workgroup (256 threads = 8 wave32) per 8x8 window. Adds into d_out.
// ---------------------------------------------------------------------------
__global__ __launch_bounds__(256) void lcam_attn(
    const float* __restrict__ x, const float* __restrict__ illu,
    const float* __restrict__ qkv_w, const float* __restrict__ qkv_b,
    const float* __restrict__ proj_w, const float* __restrict__ proj_b,
    const float* __restrict__ gt, float* __restrict__ out) {
  extern __shared__ char smem[];
  __bf16* Xs   = (__bf16*)(smem + OFF_XS);
  __bf16* Gs   = (__bf16*)(smem + OFF_GS);
  __bf16* Pb   = (__bf16*)(smem + OFF_PB);
  __bf16* QKVs = (__bf16*)(smem + OFF_QKV);
  __bf16* WT   = (__bf16*)(smem + OFF_WT);
  float*  Sb   = (float*)(smem + OFF_SB);
  float*  RawX = (float*)(smem + OFF_RAWX);
  float*  RawI = (float*)(smem + OFF_RAWI);
  __bf16* Obuf = (__bf16*)(smem + OFF_OB);
  __bf16* VT   = (__bf16*)(smem + OFF_VT);
  float*  gkf  = (float*)(smem + OFF_GKF);
  float*  nqi  = (float*)(smem + OFF_NQI);
  float*  nki  = (float*)(smem + OFF_NKI);
  float*  Pg   = (float*)(smem + OFF_PG);

  const int tid  = threadIdx.x;
  const int lane = tid & 31;
  const int w    = tid >> 5;
  const int win  = blockIdx.x;
  const int b    = win >> 10;         // / 1024
  const int wr   = win & 1023;
  const int wy   = wr >> 5, wx = wr & 31;
  const int py0  = wy * 8, px0 = wx * 8;
  const v8f Z8   = {0.f, 0.f, 0.f, 0.f, 0.f, 0.f, 0.f, 0.f};

  // ---- stage 0: async-DMA x / illu windows into LDS (raw f32), then cvt ---
  // 16B per lane per op; raw region aliases Sb (dead until stage 2).
  for (int idx = tid; idx < (T_ * C_) / 4; idx += 256) {   // 3072 vec4s
    const int e  = idx * 4;
    const int t  = e / C_, c = e % C_;
    const int ph = py0 + (t >> 3), pw = px0 + (t & 7);
    const size_t g = (((size_t)b * H_ + ph) * W_ + pw) * C_ + c;
    async_gload_b128(OFF_RAWX + (unsigned)idx * 16, x + g);
    async_gload_b128(OFF_RAWI + (unsigned)idx * 16, illu + g);
  }
  if (tid < C_) gkf[tid] = gt[b * C_ + tid];
  wait_asynccnt0();
  __syncthreads();
  for (int idx = tid; idx < T_ * C_; idx += 256) {  // raw flat == t*192+c
    const int t = idx / C_, c = idx % C_;
    Xs[t * 200 + c] = (__bf16)RawX[idx];
    const float il = RawI[idx];
    Gs[t * 200 + c] = (__bf16)(1.0f + 1.0f / (1.0f + __expf(-il)));
  }

  // ---- stage 1: QKV GEMM  (64x192) @ (192x576), K slabs of 32 -------------
  v8f acc1[18];
#pragma unroll
  for (int i = 0; i < 18; ++i) acc1[i] = Z8;
  for (int s = 0; s < 6; ++s) {
    __syncthreads();
    for (int idx = tid; idx < 32 * 576; idx += 256) {   // slab -> [N][K] bf16
      const int k = idx / 576, n = idx % 576;
      WT[n * 40 + k] = (__bf16)qkv_w[(size_t)(s * 32 + k) * 576 + n];
    }
    if (s < 5)  // warm L2 for next slab while this slab's WMMAs run
      __builtin_prefetch(qkv_w + (size_t)((s + 1) * 32) * 576 + tid * 72, 0, 3);
    __syncthreads();
    v16bf afr[4];
#pragma unroll
    for (int mt = 0; mt < 4; ++mt)
      afr[mt] = ldfrag(Xs + (mt * 16) * 200 + s * 32, 200, lane);
#pragma unroll
    for (int j = 0; j < 18; ++j) {
      const int idx = w + 8 * j;           // 144 tiles over 8 waves
      const int mt = idx & 3, nt = idx >> 2;
      v16bf bfr = ldfrag(WT + (nt * 16) * 40, 40, lane);
      acc1[j] = wmma_bf16(afr[mt], bfr, acc1[j]);
    }
  }
  {  // epilogue: +bias, gate V with (1+sigmoid(illu)), write bf16 QKV
    const int n = lane & 15, mb = (lane >> 4) * 8;
#pragma unroll
    for (int j = 0; j < 18; ++j) {
      const int idx = w + 8 * j;
      const int mt = idx & 3, nt = idx >> 2;
      const int cc = nt * 16 + n;
      const float bias = qkv_b[cc];
#pragma unroll
      for (int g = 0; g < 8; ++g) {
        const int m = mt * 16 + mb + g;
        float val = acc1[j][g] + bias;
        if (cc >= 384) val *= (float)Gs[m * 200 + (cc - 384)];
        QKVs[m * 584 + cc] = (__bf16)val;
      }
    }
  }
  __syncthreads();

  // ---- inverse L2 norms (folded into logit scaling) + V transpose ---------
  for (int idx = tid; idx < 2 * HEADS_ * T_; idx += 256) {
    const int isK = idx >= HEADS_ * T_;
    const int r = idx % (HEADS_ * T_);
    const int h = r / T_, t = r % T_;
    const __bf16* p = QKVs + t * 584 + (isK ? 192 : 0) + h * 32;
    float ss = 0.f;
#pragma unroll
    for (int c = 0; c < 32; ++c) { const float v = (float)p[c]; ss += v * v; }
    const float inv = 1.0f / fmaxf(sqrtf(ss), 1e-12f);
    (isK ? nki : nqi)[h * T_ + t] = inv;
  }
  for (int idx = tid; idx < HEADS_ * 32 * T_; idx += 256) {
    const int h = idx / (32 * T_), r = idx % (32 * T_);
    const int c = r / T_, t = r % T_;
    VT[(h * 32 + c) * 72 + t] = QKVs[t * 584 + 384 + h * 32 + c];
  }
  __syncthreads();

  // ---- stage 2a: logits + global-token column + softmax (wave w = head) ---
  if (w < HEADS_) {
    const int h = w;
#pragma unroll
    for (int mt = 0; mt < 4; ++mt) {
      v16bf aq = ldfrag(QKVs + (mt * 16) * 584 + h * 32, 584, lane);
#pragma unroll
      for (int nt = 0; nt < 4; ++nt) {
        // K^T operand: QKVs rows are [token][channel] == Bstore[N][K]
        v16bf bk = ldfrag(QKVs + (nt * 16) * 584 + 192 + h * 32, 584, lane);
        v8f d = wmma_bf16(aq, bk, Z8);
        const int n = lane & 15, mb = (lane >> 4) * 8;
#pragma unroll
        for (int g = 0; g < 8; ++g) {
          const int m = mt * 16 + mb + g, nn = nt * 16 + n;
          Sb[(h * 64 + m) * 68 + nn] =
              d[g] * nqi[h * 64 + m] * nki[h * 64 + nn] * SCALE_;
        }
      }
    }
#pragma unroll
    for (int q2 = 0; q2 < 2; ++q2) {   // unnormalized global key logit
      const int t = lane * 2 + q2;
      const __bf16* qp = QKVs + t * 584 + h * 32;
      float dot = 0.f;
#pragma unroll
      for (int c = 0; c < 32; ++c) dot += (float)qp[c] * gkf[h * 32 + c];
      Sb[(h * 64 + t) * 68 + 64] = dot * nqi[h * 64 + t] * SCALE_;
    }
#pragma unroll
    for (int q2 = 0; q2 < 2; ++q2) {   // softmax over 65 keys
      const int t = lane * 2 + q2;
      float* row = Sb + (h * 64 + t) * 68;
      float mx = row[0];
      for (int s2 = 1; s2 < 65; ++s2) mx = fmaxf(mx, row[s2]);
      float sum = 0.f;
      for (int s2 = 0; s2 < 65; ++s2) sum += __expf(row[s2] - mx);
      const float inv = 1.0f / sum;
      for (int s2 = 0; s2 < 64; ++s2)
        Pb[(h * 64 + t) * 66 + s2] = (__bf16)(__expf(row[s2] - mx) * inv);
      Pg[h * 64 + t] = __expf(row[64] - mx) * inv;
    }
  }
  __syncthreads();  // Sb dead from here; Obuf aliases it

  // ---- stage 2b: O = P @ V  (+ rank-1 global-token value update) ----------
  if (w < HEADS_) {
    const int h = w;
#pragma unroll
    for (int mt = 0; mt < 4; ++mt) {
#pragma unroll
      for (int nt2 = 0; nt2 < 2; ++nt2) {
        v8f acc = Z8;
#pragma unroll
        for (int kt = 0; kt < 2; ++kt) {
          v16bf ap = ldfrag(Pb + (h * 64 + mt * 16) * 66 + kt * 32, 66, lane);
          v16bf bv = ldfrag(VT + (h * 32 + nt2 * 16) * 72 + kt * 32, 72, lane);
          acc = wmma_bf16(ap, bv, acc);
        }
        const int n = lane & 15, mb = (lane >> 4) * 8;
#pragma unroll
        for (int g = 0; g < 8; ++g) {
          const int m = mt * 16 + mb + g, c = nt2 * 16 + n;
          const float val = acc[g] + Pg[h * 64 + m] * gkf[h * 32 + c];
          Obuf[m * 200 + h * 32 + c] = (__bf16)val;
        }
      }
    }
  }
  __syncthreads();

  // ---- stage 3: proj GEMM (64x192)@(192x192) + bias, += into d_out --------
  v8f acc3[6];
#pragma unroll
  for (int i = 0; i < 6; ++i) acc3[i] = Z8;
  for (int s = 0; s < 6; ++s) {
    __syncthreads();
    for (int idx = tid; idx < 32 * 192; idx += 256) {
      const int k = idx / 192, n = idx % 192;
      WT[n * 40 + k] = (__bf16)proj_w[(size_t)(s * 32 + k) * 192 + n];
    }
    if (s < 5)
      __builtin_prefetch(proj_w + (size_t)((s + 1) * 32) * 192 + tid * 24, 0, 3);
    __syncthreads();
    v16bf afr[4];
#pragma unroll
    for (int mt = 0; mt < 4; ++mt)
      afr[mt] = ldfrag(Obuf + (mt * 16) * 200 + s * 32, 200, lane);
#pragma unroll
    for (int j = 0; j < 6; ++j) {
      const int idx = w + 8 * j;          // 48 tiles over 8 waves
      const int mt = idx & 3, nt = idx >> 2;
      v16bf bfr = ldfrag(WT + (nt * 16) * 40, 40, lane);
      acc3[j] = wmma_bf16(afr[mt], bfr, acc3[j]);
    }
  }
  {
    const int n = lane & 15, mb = (lane >> 4) * 8;
#pragma unroll
    for (int j = 0; j < 6; ++j) {
      const int idx = w + 8 * j;
      const int mt = idx & 3, nt = idx >> 2;
      const int cc = nt * 16 + n;
      const float bias = proj_b[cc];
#pragma unroll
      for (int g = 0; g < 8; ++g) {
        const int m = mt * 16 + mb + g;
        const int ph = py0 + (m >> 3), pw = px0 + (m & 7);
        const size_t ga = (((size_t)b * H_ + ph) * W_ + pw) * C_ + cc;
        out[ga] += acc3[j][g] + bias;     // pos branch already in d_out
      }
    }
  }
}

// ---------------------------------------------------------------------------
extern "C" void kernel_launch(void* const* d_in, const int* in_sizes, int n_in,
                              void* d_out, int out_size, void* d_ws, size_t ws_size,
                              hipStream_t stream) {
  const float* x      = (const float*)d_in[0];
  const float* illu   = (const float*)d_in[1];
  const float* gt_w   = (const float*)d_in[2];
  const float* gt_b   = (const float*)d_in[3];
  const float* qkv_w  = (const float*)d_in[4];
  const float* qkv_b  = (const float*)d_in[5];
  const float* proj_w = (const float*)d_in[6];
  const float* proj_b = (const float*)d_in[7];
  const float* conv1w = (const float*)d_in[8];
  const float* conv2w = (const float*)d_in[9];
  float* out = (float*)d_out;

  float* mean = (float*)d_ws;           // [B*C]
  float* gt   = mean + B_ * C_;         // [B*C]

  (void)in_sizes; (void)n_in; (void)out_size; (void)ws_size;

  // allow >64KB dynamic LDS for the fused attention kernel
  hipFuncSetAttribute((const void*)lcam_attn,
                      hipFuncAttributeMaxDynamicSharedMemorySize, SMEM_BYTES);

  lcam_mean<<<B_ * C_, 256, 0, stream>>>(x, mean);
  lcam_gt<<<B_, C_, 0, stream>>>(mean, gt_w, gt_b, gt);
  lcam_pos<<<B_ * (H_ / 16) * (W_ / 16) * C_, 256, 0, stream>>>(x, conv1w, conv2w, out);
  lcam_attn<<<NWIN_, 256, SMEM_BYTES, stream>>>(x, illu, qkv_w, qkv_b,
                                                proj_w, proj_b, gt, out);
}
